// PositionAttentionModule_32444182954274
// MI455X (gfx1250) — matrix operand order repaired and checked by reading, not compiled, not run on hardware
//
#include <hip/hip_runtime.h>

// ---------------------------------------------------------------------------
// Position-attention (B=4, C=512, Cr=64, N=64*64=4096), MI455X / gfx1250.
// Precision: f16 operands + f32 WMMA accumulation (flash-attention standard).
// Workspace (f16): qT[B][N][64] | kT[B][N][64] | v[B][C][N] | Wqh | Wkh | Wvh
// ---------------------------------------------------------------------------

typedef __attribute__((ext_vector_type(16))) _Float16 v16h;
typedef __attribute__((ext_vector_type(8)))  _Float16 v8h;
typedef __attribute__((ext_vector_type(8)))  float    v8f;

#define B_  4
#define C_  512
#define CR_ 64
#define N_  4096

union V16 { v16h v; v8h h[2]; };

static __device__ __forceinline__ v8f wmma_f16(v16h a, v16h b, v8f c) {
  // v_wmma_f32_16x16x32_f16  (D = A[16x32] * B[32x16] + C[16x16], f32 acc)
  return __builtin_amdgcn_wmma_f32_16x16x32_f16(
      /*neg_a=*/false, a, /*neg_b=*/false, b,
      /*c_mod=*/(short)0, c, /*reuse_a=*/false, /*reuse_b=*/false);
}

// A-operand (16xK rows, f16): lane holds row M = lane%16.
// lanes 0-15: K = {k0..k0+7, k0+16..k0+23}; lanes 16-31: +8.  (ISA 7.12.2)
static __device__ __forceinline__ v16h load_a(const _Float16* row, int kbase, int lhalf) {
  V16 u;
  u.h[0] = *(const v8h*)(row + kbase + lhalf * 8);
  u.h[1] = *(const v8h*)(row + kbase + lhalf * 8 + 16);
  return u.v;
}

// B-operand (Kx16 cols, f16): lane holds column N = lane%16;
// lanes 0-15: K = {0..7, 16..23}; lanes 16-31: +8 (matches load_a).
static __device__ __forceinline__ v16h load_b(const _Float16* col, int lhalf) {
  V16 u;
  u.h[0] = *(const v8h*)(col + lhalf * 8);
  u.h[1] = *(const v8h*)(col + lhalf * 8 + 16);
  return u.v;
}

// ---------------------------------------------------------------------------
// Stage 0: one-shot f32 -> f16 conversion of the weight matrices so the
// stage-1 GEMM inner loops are pure loads + WMMA (no per-iteration v_cvt).
// ---------------------------------------------------------------------------
__global__ __launch_bounds__(256) void cvt_weights(
    const float* __restrict__ Wq, const float* __restrict__ Wk,
    const float* __restrict__ Wv,
    _Float16* __restrict__ Wqh, _Float16* __restrict__ Wkh,
    _Float16* __restrict__ Wvh)
{
  const int i = blockIdx.x * 256 + threadIdx.x;       // grid covers 262144
  if (i < CR_ * C_) {                                 // 32768
    Wqh[i] = (_Float16)Wq[i];
    Wkh[i] = (_Float16)Wk[i];
  }
  Wvh[i] = (_Float16)Wv[i];                           // 262144 elements
}

// ---------------------------------------------------------------------------
// Stage 1: q/k/v 1x1-conv projections (GEMMs over K=C=512).
// One workgroup = one batch x one 32-wide slice of positions n.
// x slice is transposed+converted into LDS once, reused by all 80 WMMA jobs.
// ---------------------------------------------------------------------------
__global__ __launch_bounds__(256) void qkv_proj(
    const float* __restrict__ x,
    const _Float16* __restrict__ Wqh, const float* __restrict__ bq,
    const _Float16* __restrict__ Wkh, const float* __restrict__ bk,
    const _Float16* __restrict__ Wvh, const float* __restrict__ bv,
    _Float16* __restrict__ qT, _Float16* __restrict__ kT,
    _Float16* __restrict__ vO)
{
  __shared__ __align__(32) _Float16 xs[32][520];   // [n_local][c], padded rows

  const int b  = blockIdx.x >> 7;                  // N_/32 = 128 tiles
  const int n0 = (blockIdx.x & 127) * 32;
  const int t  = threadIdx.x;
  const float* xb = x + (size_t)b * C_ * N_;

  {  // coalesced read of x[c, n0..n0+31], transposed f16 store into LDS
    const int n_l = t & 31;
    for (int c = t >> 5; c < C_; c += 8)
      xs[n_l][c] = (_Float16)xb[(size_t)c * N_ + n0 + n_l];
  }
  __syncthreads();

  const int wave = t >> 5, lane = t & 31;
  const int lhalf = lane >> 4, l16 = lane & 15;

  // 80 jobs = 2 n-subtiles x (4 q-tiles + 4 k-tiles + 32 v-tiles)
  for (int job = wave; job < 80; job += 8) {
    const int n_sub = job & 1;
    const int ot    = job >> 1;
    v8f acc = {};

    if (ot < 8) {
      // qT/kT: D tile M=n(16) x N=o(16), K=c.  B cols = rows of Wqh/Wkh.
      const bool isq = (ot < 4);
      const int  o   = ((ot & 3) << 4) + l16;
      const _Float16* wrow = (isq ? Wqh : Wkh) + (size_t)o * C_;
      const _Float16* arow = &xs[(n_sub << 4) + l16][0];
      #pragma unroll 4
      for (int cb = 0; cb < C_; cb += 32)
        acc = wmma_f16(load_a(arow, cb, lhalf), load_b(wrow + cb, lhalf), acc);
      const float bias = isq ? bq[o] : bk[o];
      _Float16* dst = isq ? qT : kT;
      #pragma unroll
      for (int r = 0; r < 8; ++r) {
        const int n_g = n0 + (n_sub << 4) + r + (lhalf << 3);
        dst[((size_t)b * N_ + n_g) * CR_ + o] = (_Float16)(acc[r] + bias);
      }
    } else {
      // v: D tile M=c_out(16) x N=n(16), K=c_in.  A rows = rows of Wvh.
      const int ct = ot - 8;
      const _Float16* arow = Wvh + (size_t)((ct << 4) + l16) * C_;
      const _Float16* bcol = &xs[(n_sub << 4) + l16][0];
      #pragma unroll 4
      for (int cb = 0; cb < C_; cb += 32)
        acc = wmma_f16(load_a(arow, cb, lhalf), load_b(bcol + cb, lhalf), acc);
      #pragma unroll
      for (int r = 0; r < 8; ++r) {
        const int c_g = (ct << 4) + r + (lhalf << 3);
        const int n_g = n0 + (n_sub << 4) + l16;
        vO[((size_t)b * C_ + c_g) * N_ + n_g] = (_Float16)(acc[r] + bv[c_g]);
      }
    }
  }
}

// ---------------------------------------------------------------------------
// Stage 2: fused attention, flash-style over j-tiles of 64.
// One workgroup = one batch x 32 query rows. 8 waves:
//   phase A: wave (iw,jw) computes a 16x16 S subtile (2 chained WMMAs, K=64)
//   phase B: 32 row-owner threads do online softmax, P(f16) -> LDS
//   phase C: wave w owns output channels [64w,64w+64): 16 WMMAs of P*V^T
// ---------------------------------------------------------------------------
__global__ __launch_bounds__(256) void attn_fused(
    const _Float16* __restrict__ qT, const _Float16* __restrict__ kT,
    const _Float16* __restrict__ vO, const float* __restrict__ x,
    const float* __restrict__ gamma, float* __restrict__ out)
{
  __shared__ float Ssc[32][65];                    // raw scores tile (padded)
  __shared__ __align__(32) _Float16 Ps[32][72];    // softmaxed probs, A-ready rows
  __shared__ float rowscale[32];
  __shared__ float rowsum[32];

  const int b  = blockIdx.x >> 7;
  const int i0 = (blockIdx.x & 127) * 32;
  const int t  = threadIdx.x;
  const int wave = t >> 5, lane = t & 31;
  const int lhalf = lane >> 4, l16 = lane & 15;
  const int iw = wave >> 2;        // i-subtile for phase A (0..1)
  const int jw = wave & 3;         // j-subtile for phase A (0..3)
  const int c0 = wave << 6;        // this wave's 64-channel output slice

  // q A-operand is loop-invariant: load once.
  const _Float16* qrow = qT + ((size_t)b * N_ + i0 + (iw << 4) + l16) * CR_;
  const v16h Aq0 = load_a(qrow, 0, lhalf);
  const v16h Aq1 = load_a(qrow, 32, lhalf);

  float m_run = -INFINITY, l_run = 0.0f;
  v8f acc[2][4] = {};              // [i_sub][c_sub] 16x16 f32 accumulators

  for (int j0 = 0; j0 < N_; j0 += 64) {
    // ---- phase A: S = q^T k ----
    const _Float16* kcol = kT + ((size_t)b * N_ + j0 + (jw << 4) + l16) * CR_;
    v8f s = {};
    s = wmma_f16(Aq0, load_b(kcol, lhalf), s);
    s = wmma_f16(Aq1, load_b(kcol + 32, lhalf), s);
    #pragma unroll
    for (int r = 0; r < 8; ++r)
      Ssc[(iw << 4) + r + (lhalf << 3)][(jw << 4) + l16] = s[r];
    __syncthreads();

    // ---- phase B: online softmax per row ----
    if (t < 32) {
      float mx = m_run;
      for (int jj = 0; jj < 64; ++jj) mx = fmaxf(mx, Ssc[t][jj]);
      const float sc = __expf(m_run - mx);   // exp(-inf)=0 on first tile
      float ls = l_run * sc;
      for (int jj = 0; jj < 64; ++jj) {
        const float p = __expf(Ssc[t][jj] - mx);
        ls += p;
        Ps[t][jj] = (_Float16)p;
      }
      m_run = mx; l_run = ls;
      rowscale[t] = sc;
    }
    __syncthreads();

    // ---- rescale accumulators by exp(m_old - m_new) ----
    #pragma unroll
    for (int is = 0; is < 2; ++is)
      #pragma unroll
      for (int r = 0; r < 8; ++r) {
        const float sc = rowscale[(is << 4) + r + (lhalf << 3)];
        #pragma unroll
        for (int cs = 0; cs < 4; ++cs) acc[is][cs][r] *= sc;
      }

    // ---- phase C: acc += P * V^T (this wave's channel slice) ----
    v16h Pa[2][2];
    #pragma unroll
    for (int is = 0; is < 2; ++is)
      #pragma unroll
      for (int jc = 0; jc < 2; ++jc)
        Pa[is][jc] = load_a(&Ps[(is << 4) + l16][0], jc * 32, lhalf);
    #pragma unroll
    for (int cs = 0; cs < 4; ++cs) {
      const _Float16* vcol =
          vO + ((size_t)b * C_ + c0 + (cs << 4) + l16) * N_ + j0;
      const v16h Bv0 = load_b(vcol, lhalf);
      const v16h Bv1 = load_b(vcol + 32, lhalf);
      #pragma unroll
      for (int is = 0; is < 2; ++is) {
        acc[is][cs] = wmma_f16(Pa[is][0], Bv0, acc[is][cs]);
        acc[is][cs] = wmma_f16(Pa[is][1], Bv1, acc[is][cs]);
      }
    }

    // hint the next k/v tiles toward the caches (global_prefetch_b8)
    if (j0 + 64 < N_) {
      __builtin_prefetch(kT + ((size_t)b * N_ + j0 + 64 + (jw << 4) + l16) * CR_, 0, 1);
      __builtin_prefetch(vO + ((size_t)b * C_ + c0 + l16) * N_ + j0 + 64, 0, 1);
    }
    __syncthreads();   // protects Ps/Ssc reads vs next iteration's writers
  }

  // ---- epilogue: out = gamma * acc / l + x ----
  if (t < 32) rowsum[t] = l_run;
  __syncthreads();
  const float g = gamma[0];
  #pragma unroll
  for (int is = 0; is < 2; ++is)
    #pragma unroll
    for (int cs = 0; cs < 4; ++cs) {
      const int cg = c0 + (cs << 4) + l16;
      #pragma unroll
      for (int r = 0; r < 8; ++r) {
        const int i_loc = (is << 4) + r + (lhalf << 3);
        const size_t idx = ((size_t)b * C_ + cg) * (size_t)N_ + (size_t)(i0 + i_loc);
        out[idx] = g * (acc[is][cs][r] / rowsum[i_loc]) + x[idx];
      }
    }
}

// ---------------------------------------------------------------------------
extern "C" void kernel_launch(void* const* d_in, const int* in_sizes, int n_in,
                              void* d_out, int out_size, void* d_ws, size_t ws_size,
                              hipStream_t stream) {
  (void)in_sizes; (void)n_in; (void)out_size; (void)ws_size;
  const float* x     = (const float*)d_in[0];
  const float* Wq    = (const float*)d_in[1];
  const float* bq    = (const float*)d_in[2];
  const float* Wk    = (const float*)d_in[3];
  const float* bk    = (const float*)d_in[4];
  const float* Wv    = (const float*)d_in[5];
  const float* bv    = (const float*)d_in[6];
  const float* gamma = (const float*)d_in[7];
  float* out = (float*)d_out;

  // f16 workspace: qT | kT | v | Wqh | Wkh | Wvh
  _Float16* ws  = (_Float16*)d_ws;
  _Float16* qT  = ws;
  _Float16* kT  = qT + (size_t)B_ * N_ * CR_;        // +1,048,576
  _Float16* vO  = kT + (size_t)B_ * N_ * CR_;        // +1,048,576
  _Float16* Wqh = vO + (size_t)B_ * C_ * N_;         // +8,388,608
  _Float16* Wkh = Wqh + (size_t)CR_ * C_;            // +32,768
  _Float16* Wvh = Wkh + (size_t)CR_ * C_;            // +32,768 (Wvh: 262,144)

  cvt_weights<<<(C_ * C_) / 256, 256, 0, stream>>>(Wq, Wk, Wv, Wqh, Wkh, Wvh);
  qkv_proj<<<B_ * (N_ / 32), 256, 0, stream>>>(x, Wqh, bq, Wkh, bk, Wvh, bv, qT, kT, vO);
  attn_fused<<<B_ * (N_ / 32), 256, 0, stream>>>(qT, kT, vO, x, gamma, out);
}
